// SRULSTMCell_54150947668389
// MI455X (gfx1250) — compile-verified, matrix operand order checked by reading, not run on hardware
//
#include <hip/hip_runtime.h>

// SRU-LSTM cell fused kernel for gfx1250 (MI455X), bf16 WMMA path.
// B=4096, IN_DIM=1024, HID=1024.

#define HID    1024
#define NROWS  4096

#define BM   128      // rows per workgroup tile
#define BN   32       // output columns per workgroup tile (2 x 16 subtiles)
#define BK   32       // K step (matches 16x16x32 bf16 WMMA)
#define ASTR 40       // LDS row pitch in ushort (80B: 16B-aligned, 20 banks)
#define TPB  256      // 8 waves of 32

typedef __attribute__((ext_vector_type(16))) __bf16 bf16x16;
typedef __attribute__((ext_vector_type(8)))  __bf16 bf16x8;
typedef __attribute__((ext_vector_type(8)))  float  floatx8;

union Frag16 { bf16x16 v; bf16x8 h[2]; };

// Pack two f32 -> packed bf16x2 (round-to-nearest-even).
__device__ __forceinline__ unsigned pack2(float a, float b) {
#if __has_builtin(__builtin_amdgcn_cvt_pk_bf16_f32)
  auto p = __builtin_amdgcn_cvt_pk_bf16_f32(a, b);   // lo=a, hi=b
  return __builtin_bit_cast(unsigned, p);
#else
  unsigned ua = __float_as_uint(a);
  ua += 0x7FFFu + ((ua >> 16) & 1u);
  unsigned ub = __float_as_uint(b);
  ub += 0x7FFFu + ((ub >> 16) & 1u);
  return (ua >> 16) | (ub & 0xFFFF0000u);
#endif
}
__device__ __forceinline__ float sigmoidf_(float x) {
  return 1.0f / (1.0f + __expf(-x));
}
__device__ __forceinline__ float tanhf_(float x) {
  float t = __expf(-2.0f * fabsf(x));
  float r = (1.0f - t) / (1.0f + t);
  return copysignf(r, x);
}

__device__ __forceinline__ void cvt16(const float4* __restrict__ s, uint4* dst) {
  float4 f0 = s[0], f1 = s[1], f2 = s[2], f3 = s[3];
  uint4 p0, p1;
  p0.x = pack2(f0.x, f0.y); p0.y = pack2(f0.z, f0.w);
  p0.z = pack2(f1.x, f1.y); p0.w = pack2(f1.z, f1.w);
  p1.x = pack2(f2.x, f2.y); p1.y = pack2(f2.z, f2.w);
  p1.z = pack2(f3.x, f3.y); p1.w = pack2(f3.z, f3.w);
  dst[0] = p0; dst[1] = p1;
}

// One K=1024 sweep. PHASE 0: A = x, gates {Wi,Wf,Wo,Wxs,Wxg} (cols 0..1023).
// PHASE 1: A = h_prev, gates {Wi,Wf,Wo (cols 1024..2047), Whg}; 's' gate skipped.
template<int PHASE>
__device__ __forceinline__ void gemm_phase(
    const float* __restrict__ Asrc,
    const float* __restrict__ Wi, const float* __restrict__ Wf,
    const float* __restrict__ Wo, const float* __restrict__ W3,
    const float* __restrict__ W4,
    unsigned short* sA, unsigned short* sB,
    int m0, int n0, int tid, int wave, int l16, int lhalf,
    floatx8 acc[5][2])
{
  const int NMAT = PHASE ? 4 : 5;
  const int ar = tid >> 1;                 // 0..127 : A tile row
  const int ah = tid & 1;                  // half-row (16 floats)
  const float* asrc = Asrc + (size_t)(m0 + ar) * HID + ah * 16;

  for (int k0 = 0; k0 < 1024; k0 += BK) {
    // ---- stage A tile: 128 x 32 f32 -> bf16 in LDS ----
    cvt16((const float4*)(asrc + k0), (uint4*)&sA[ar * ASTR + ah * 16]);

    // ---- stage B tiles: NMAT x (32 rows x 32 K) f32 -> bf16 in LDS ----
    for (int task = tid; task < NMAT * 64; task += TPB) {
      const int g   = task >> 6;           // gate index
      const int idx = task & 63;
      const int r   = idx >> 1;            // 0..31 : output column within tile
      const int h   = idx & 1;
      const float* wsrc; int ld; int colb;
      if (PHASE == 0) {
        colb = 0;
        if      (g == 0) { wsrc = Wi; ld = 2048; }
        else if (g == 1) { wsrc = Wf; ld = 2048; }
        else if (g == 2) { wsrc = Wo; ld = 2048; }
        else if (g == 3) { wsrc = W3; ld = 1024; }   // Wxs
        else             { wsrc = W4; ld = 1024; }   // Wxg
      } else {
        if      (g == 0) { wsrc = Wi; ld = 2048; colb = 1024; }
        else if (g == 1) { wsrc = Wf; ld = 2048; colb = 1024; }
        else if (g == 2) { wsrc = Wo; ld = 2048; colb = 1024; }
        else             { wsrc = W3; ld = 1024; colb = 0;    } // Whg
      }
      cvt16((const float4*)(wsrc + (size_t)(n0 + r) * ld + colb + k0 + h * 16),
            (uint4*)&sB[(g * BN + r) * ASTR + h * 16]);
    }
    __syncthreads();

    // ---- fragments + WMMA (software-pipelined B-fragment loads) ----
    // A-frag (16x32 bf16): lane holds row M=l16; K = lhalf*8+{0..7}, 16+lhalf*8+{0..7}
    Frag16 a;
    const int arow = (wave * 16 + l16) * ASTR;
    a.h[0] = *(const bf16x8*)&sA[arow + lhalf * 8];
    a.h[1] = *(const bf16x8*)&sA[arow + 16 + lhalf * 8];

    Frag16 bb[2];
    {
      const int brow0 = (0 * BN + 0 * 16 + l16) * ASTR;
      bb[0].h[0] = *(const bf16x8*)&sB[brow0 + lhalf * 8];
      bb[0].h[1] = *(const bf16x8*)&sB[brow0 + 16 + lhalf * 8];
    }
#pragma unroll
    for (int idx = 0; idx < 2 * NMAT; ++idx) {
      if (idx + 1 < 2 * NMAT) {
        const int gn   = (idx + 1) >> 1;
        const int subn = (idx + 1) & 1;
        const int brow = (gn * BN + subn * 16 + l16) * ASTR;
        bb[(idx + 1) & 1].h[0] = *(const bf16x8*)&sB[brow + lhalf * 8];
        bb[(idx + 1) & 1].h[1] = *(const bf16x8*)&sB[brow + 16 + lhalf * 8];
      }
      const int g   = idx >> 1;
      const int sub = idx & 1;
      const int ai  = (PHASE && g == 3) ? 4 : g;  // phase1 g==3 is Whg -> cand acc
      acc[ai][sub] = __builtin_amdgcn_wmma_f32_16x16x32_bf16(
          false, a.v, false, bb[idx & 1].v, (short)0, acc[ai][sub], false, false);
    }
    __syncthreads();
  }
}

__global__ __launch_bounds__(TPB) void srulstm_wmma_kernel(
    const float* __restrict__ x, const float* __restrict__ h_prev,
    const float* __restrict__ c_prev,
    const float* __restrict__ Wi,  const float* __restrict__ bi,
    const float* __restrict__ Wf,  const float* __restrict__ bf_,
    const float* __restrict__ Wo,  const float* __restrict__ bo,
    const float* __restrict__ Wxs, const float* __restrict__ bxs,
    const float* __restrict__ Wxg, const float* __restrict__ bxg,
    const float* __restrict__ Whg, const float* __restrict__ bhg,
    float* __restrict__ out_h, float* __restrict__ out_c)
{
  __shared__ __align__(16) unsigned short sA[BM * ASTR];       // 10.0 KB
  __shared__ __align__(16) unsigned short sB[5 * BN * ASTR];   // 12.5 KB

  const int tid   = threadIdx.x;
  const int wave  = tid >> 5;
  const int lane  = tid & 31;
  const int l16   = lane & 15;
  const int lhalf = lane >> 4;

  const int m0 = blockIdx.x * BM;
  const int n0 = blockIdx.y * BN;

  // acc[gate][n-subtile]; gate: 0=i 1=f 2=o 3=s 4=cand
  floatx8 acc[5][2];
#pragma unroll
  for (int i = 0; i < 5; ++i)
#pragma unroll
    for (int s = 0; s < 2; ++s)
      acc[i][s] = (floatx8)(0.0f);

  gemm_phase<0>(x,      Wi, Wf, Wo, Wxs, Wxg, sA, sB, m0, n0, tid, wave, l16, lhalf, acc);
  gemm_phase<1>(h_prev, Wi, Wf, Wo, Whg, Whg, sA, sB, m0, n0, tid, wave, l16, lhalf, acc);

  // ---- fused epilogue, all in registers ----
  // C/D layout: c[j] -> M = j + 8*lhalf, N = l16 (per 16x16 subtile)
#pragma unroll
  for (int sub = 0; sub < 2; ++sub) {
    const int n = n0 + sub * 16 + l16;
    const float Bi = bi[n];
    const float Bf = bf_[n];
    const float Bo = bo[n];
    const float Bs = bxs[n];
    const float Bc = bxg[n] + bhg[n];
    const int mbase = m0 + wave * 16 + lhalf * 8;
#pragma unroll
    for (int j = 0; j < 8; ++j) {
      const size_t off = (size_t)(mbase + j) * HID + n;
      const float iv = sigmoidf_(acc[0][sub][j] + Bi);
      const float fv = sigmoidf_(acc[1][sub][j] + Bf);
      const float ov = sigmoidf_(acc[2][sub][j] + Bo);
      const float sv = acc[3][sub][j] + Bs;
      const float cv = acc[4][sub][j] + Bc;
      const float g  = tanhf_(sv * cv);
      const float c  = fv * c_prev[off] + iv * g;
      out_c[off] = c;
      out_h[off] = ov * tanhf_(c);
    }
  }
}

extern "C" void kernel_launch(void* const* d_in, const int* in_sizes, int n_in,
                              void* d_out, int out_size, void* d_ws, size_t ws_size,
                              hipStream_t stream) {
  (void)in_sizes; (void)n_in; (void)out_size; (void)d_ws; (void)ws_size;
  const float* x      = (const float*)d_in[0];
  const float* h_prev = (const float*)d_in[1];
  const float* c_prev = (const float*)d_in[2];
  const float* Wi  = (const float*)d_in[3];
  const float* bi  = (const float*)d_in[4];
  const float* Wf  = (const float*)d_in[5];
  const float* bf  = (const float*)d_in[6];
  const float* Wo  = (const float*)d_in[7];
  const float* bo  = (const float*)d_in[8];
  const float* Wxs = (const float*)d_in[9];
  const float* bxs = (const float*)d_in[10];
  const float* Wxg = (const float*)d_in[11];
  const float* bxg = (const float*)d_in[12];
  const float* Whg = (const float*)d_in[13];
  const float* bhg = (const float*)d_in[14];

  float* out_h = (float*)d_out;
  float* out_c = out_h + (size_t)NROWS * HID;

  dim3 grid(NROWS / BM, HID / BN);   // 32 x 32 workgroups
  srulstm_wmma_kernel<<<grid, dim3(TPB), 0, stream>>>(
      x, h_prev, c_prev, Wi, bi, Wf, bf, Wo, bo,
      Wxs, bxs, Wxg, bxg, Whg, bhg, out_h, out_c);
}
